// NMS_8856222564617
// MI455X (gfx1250) — compile-verified, matrix operand order checked by reading, not run on hardware
//
#include <hip/hip_runtime.h>
#include <stdint.h>

// ---------------------------------------------------------------------------
// Multiclass NMS for MI455X (gfx1250, wave32).
//   B=8, N=8192, C=32, PRE_TOPK=2048, MAX_DET=300, SCORE_THR=IOU_THR=0.5
//
// Phase 1: per-batch exact top-2048 via 64-bit radix-select + LDS bitonic sort
// Phase 2: 2048x2048 suppression bitmask; same-class predicate via
//          v_wmma_f32_16x16x32_f16 (one-hot class GEMM), area-sum via
//          v_wmma_f32_16x16x4_f32 (rank-2 outer product); IoU predicate is
//          division-free (inter > thr * union); lower-triangle tiles fast-path
// Phase 3: sequential greedy resolve; mask rows streamed into LDS with
//          global_load_async_to_lds_b128 + s_wait_asynccnt double buffering
// ---------------------------------------------------------------------------

#define B_      8
#define N_      8192
#define C_      32
#define NC_     (N_ * C_)
#define K_      2048
#define CT_     128          // K_/16 column tiles
#define MAXDET_ 300
#define THR_    0.5f

typedef _Float16 v16h __attribute__((ext_vector_type(16)));
typedef float    v8f  __attribute__((ext_vector_type(8)));
typedef float    v2f  __attribute__((ext_vector_type(2)));

// ===========================================================================
// Kernel 1: top-2048 selection + sort (one 1024-thread workgroup per batch)
// ===========================================================================
__global__ __launch_bounds__(1024) void nms_topk(
    const float* __restrict__ boxes,    // [B, N, 4]
    const float* __restrict__ scores,   // [B, N, C]
    float* __restrict__ cscore,         // [B, K]
    int*   __restrict__ ccls,           // [B, K]
    float* __restrict__ cbox)           // [B, K, 4]
{
  __shared__ unsigned long long skey[K_];
  __shared__ unsigned hist[256];
  __shared__ unsigned long long spref;
  __shared__ int sK;
  __shared__ unsigned scnt;

  const int b   = blockIdx.x;
  const int tid = threadIdx.x;
  const float* sc = scores + (size_t)b * NC_;

  if (tid == 0) { spref = 0ull; sK = K_; scnt = 0u; }
  for (int i = tid; i < K_; i += 1024) skey[i] = 0ull;
  __syncthreads();

  // ---- 8-round byte radix-select for the 2048th largest 64-bit key ----
  for (int rnd = 0; rnd < 8; ++rnd) {
    const int shift = 56 - 8 * rnd;
    if (tid < 256) hist[tid] = 0u;
    __syncthreads();
    const unsigned long long pref = spref;
    for (int f = tid; f < NC_; f += 1024) {
      float s = sc[f];
      if (!(s > THR_)) continue;
      unsigned long long key =
          ((unsigned long long)__float_as_uint(s) << 32) | (unsigned)(~(unsigned)f);
      if (rnd > 0 && (key >> (shift + 8)) != (pref >> (shift + 8))) continue;
      atomicAdd(&hist[(unsigned)((key >> shift) & 0xFFull)], 1u);
    }
    __syncthreads();
    if (tid == 0) {
      int Krem = sK;
      unsigned cum = 0;
      int pick = 0;
      for (int bkt = 255; bkt >= 0; --bkt) {
        if ((int)(cum + hist[bkt]) >= Krem) { pick = bkt; break; }
        cum += hist[bkt];
      }
      spref |= ((unsigned long long)pick << shift);
      sK = Krem - (int)cum;
    }
    __syncthreads();
  }
  const unsigned long long T = spref;

  // ---- gather all keys >= T (exactly 2048 when >=2048 valid; keys unique) --
  for (int f = tid; f < NC_; f += 1024) {
    float s = sc[f];
    if (!(s > THR_)) continue;
    unsigned long long key =
        ((unsigned long long)__float_as_uint(s) << 32) | (unsigned)(~(unsigned)f);
    if (key >= T && key != 0ull) {
      unsigned pos = atomicAdd(&scnt, 1u);
      if (pos < K_) skey[pos] = key;
    }
  }
  __syncthreads();

  // ---- bitonic sort, descending (score desc, flat index asc on ties) ----
  for (unsigned k = 2; k <= K_; k <<= 1) {
    for (unsigned j = k >> 1; j > 0; j >>= 1) {
      for (unsigned i = tid; i < K_; i += 1024) {
        unsigned ij = i ^ j;
        if (ij > i) {
          unsigned long long a = skey[i], c = skey[ij];
          bool descBlock = ((i & k) == 0);
          bool doswap = descBlock ? (a < c) : (a > c);
          if (doswap) { skey[i] = c; skey[ij] = a; }
        }
      }
      __syncthreads();
    }
  }

  // ---- emit candidates ----
  const float4* bx4 = (const float4*)boxes + (size_t)b * N_;
  float4*       cb4 = (float4*)cbox + (size_t)b * K_;
  for (int i = tid; i < K_; i += 1024) {
    unsigned long long key = skey[i];
    size_t o = (size_t)b * K_ + i;
    if (key == 0ull) {
      cscore[o] = -1.0f;
      ccls[o]   = 0;
      float4 z; z.x = z.y = z.z = z.w = 0.0f;
      cb4[i] = z;
    } else {
      unsigned bits = (unsigned)(key >> 32);
      unsigned flat = ~(unsigned)(key & 0xFFFFFFFFull);
      cscore[o] = __uint_as_float(bits);
      ccls[o]   = (int)(flat & (C_ - 1));
      cb4[i]    = bx4[flat >> 5];               // flat / C_
    }
  }
}

// ===========================================================================
// Kernel 2: 16x16-tile suppression bitmask. One wave per tile.
//   same-class mask  : v_wmma_f32_16x16x32_f16 on one-hot(32) class vectors
//   area_i + area_j  : v_wmma_f32_16x16x4_f32 rank-2 outer product (guarded)
//   IoU predicate    : division-free (inter > thr * max(union, eps))
//   lower triangle   : zero-store fast path (wave-uniform, skips WMMA+loads)
// ===========================================================================
__global__ __launch_bounds__(256) void nms_pairmask(
    const int*   __restrict__ ccls,     // [B, K]
    const float* __restrict__ cbox,     // [B, K, 4]
    unsigned short* __restrict__ mask)  // [B, K, CT_]
{
  const int lane = threadIdx.x & 31;
  const int wid  = threadIdx.x >> 5;
  const int t    = blockIdx.x * 8 + wid;   // tile id 0..16383
  const int b    = blockIdx.y;
  const int trow = t >> 7, tcol = t & 127;
  const int R  = trow << 4, Cc = tcol << 4;
  const int ln = lane & 15, hi = lane >> 4;

  // strictly-lower-triangle tiles: col > row never holds -> store zeros, done.
  if (tcol < trow) {
#pragma unroll
    for (int e = 0; e < 8; ++e) {
      if (lane == 0)
        mask[((size_t)b * K_ + (R + e)) * CT_ + tcol] = (unsigned short)0;
      else if (lane == 16)
        mask[((size_t)b * K_ + (R + 8 + e)) * CT_ + tcol] = (unsigned short)0;
    }
    return;
  }

  const int*    clsb = ccls + (size_t)b * K_;
  const float4* bx   = (const float4*)cbox + (size_t)b * K_;

  const int aCls = clsb[R + ln];
  const int bCls = clsb[Cc + ln];

  // One-hot(32) in f16 following the 16-bit A/B-matrix K striping:
  // lanes 0-15: VGPR r -> K = {0,1 / 2,3 / 4,5 / 6,7 / 16,17 / 18,19 / 20,21 / 22,23}
  // lanes 16-31: same +8
  v16h av, bv;
#pragma unroll
  for (int h = 0; h < 16; ++h) {
    int r  = h >> 1;
    int kk = ((r < 4) ? (2 * r) : (16 + 2 * (r - 4))) + 8 * hi + (h & 1);
    av[h] = (aCls == kk) ? (_Float16)1.0f : (_Float16)0.0f;
    bv[h] = (bCls == kk) ? (_Float16)1.0f : (_Float16)0.0f;
  }
  v8f zc = {};
  // same[m][n] = 1.0 iff cls[R+m] == cls[Cc+n]
  v8f same = __builtin_amdgcn_wmma_f32_16x16x32_f16(
      false, av, false, bv, (short)0, zc, false, false);

  const float4 cb    = bx[Cc + ln];
  const float  areaC = (cb.z - cb.x) * (cb.w - cb.y);

#if __has_builtin(__builtin_amdgcn_wmma_f32_16x16x4_f32)
#define NMS_HAS_WMMA4 1
  // asum[m][n] = area_row[m]*1 + 1*area_col[n]  (K=0,1 used; K=2,3 zero pad)
  const float4 rbl    = bx[R + ln];
  const float  areaAl = (rbl.z - rbl.x) * (rbl.w - rbl.y);
  v2f a2, b2;
  a2[0] = hi ? 0.0f : areaAl; a2[1] = hi ? 0.0f : 1.0f;
  b2[0] = hi ? 0.0f : 1.0f;   b2[1] = hi ? 0.0f : areaC;
  v8f asum = __builtin_amdgcn_wmma_f32_16x16x4_f32(
      false, a2, false, b2, (short)0, zc, false, false);
#endif

#pragma unroll
  for (int e = 0; e < 8; ++e) {
    const int    grow = R + e + 8 * hi;         // C/D layout: row = e + 8*(lane>=16)
    const float4 rb   = bx[grow];
    float ltx = fmaxf(rb.x, cb.x), lty = fmaxf(rb.y, cb.y);
    float rx  = fminf(rb.z, cb.z), ry  = fminf(rb.w, cb.w);
    float iw = fmaxf(rx - ltx, 0.0f), ih = fmaxf(ry - lty, 0.0f);
    float inter = iw * ih;
#ifdef NMS_HAS_WMMA4
    float uni = asum[e] - inter;
#else
    float areaR = (rb.z - rb.x) * (rb.w - rb.y);
    float uni = areaR + areaC - inter;
#endif
    // iou > thr  <=>  inter > thr * max(uni, eps)   (both sides non-negative)
    bool cond = (inter > THR_ * fmaxf(uni, 1e-9f)) &&
                (same[e] > 0.5f) && ((Cc + ln) > grow);

    unsigned bal = __builtin_amdgcn_ballot_w32(cond);
    if (lane == 0)
      mask[((size_t)b * K_ + (R + e)) * CT_ + tcol] = (unsigned short)(bal & 0xFFFFu);
    else if (lane == 16)
      mask[((size_t)b * K_ + (R + 8 + e)) * CT_ + tcol] = (unsigned short)(bal >> 16);
  }
}

// ===========================================================================
// Kernel 3: sequential greedy NMS resolve + output. One 256-thread WG / batch.
// Mask rows double-buffered into LDS via async-to-LDS loads (CDNA5 path).
// ===========================================================================
__global__ __launch_bounds__(256) void nms_greedy(
    const float* __restrict__ cscore,
    const int*   __restrict__ ccls,
    const float* __restrict__ cbox,
    const unsigned short* __restrict__ mask,
    float* __restrict__ outb,   // [B,300,4]
    float* __restrict__ outs,   // [B,300]
    int*   __restrict__ outl,   // [B,300]
    int*   __restrict__ outnv)  // [B,1]
{
  __shared__ unsigned removed[CT_];   // 16 valid bits per entry: column bitset
  __shared__ unsigned rowbuf[2][64];  // 256B mask row, double buffered
  __shared__ int   sflag;
  __shared__ int   skc;               // stored kept (<=300)
  __shared__ short kidx[MAXDET_];

  const int b   = blockIdx.x;
  const int tid = threadIdx.x;
  const float*          scb = cscore + (size_t)b * K_;
  const unsigned short* mb  = mask + (size_t)b * K_ * CT_;

  if (tid < CT_) removed[tid] = 0u;
  if (tid == 0) { skc = 0; sflag = 0; }
  __syncthreads();

  // prime the pipeline: async-load mask row 0 into buffer 0
  if (tid < 16) {
    unsigned    ldsoff = (unsigned)(uintptr_t)(&rowbuf[0][0]) + (unsigned)tid * 16u;
    const char* g      = (const char*)mb + (size_t)tid * 16u;
    asm volatile("global_load_async_to_lds_b128 %0, %1, off"
                 :: "v"(ldsoff), "v"(g) : "memory");
  }

  for (int i = 0; i < K_; ++i) {
    // issue async load of the next row (clamped) into the other buffer
    const int nr = (i + 1 < K_) ? (i + 1) : (K_ - 1);
    if (tid < 16) {
      unsigned    ldsoff = (unsigned)(uintptr_t)(&rowbuf[(i + 1) & 1][0]) + (unsigned)tid * 16u;
      const char* g      = (const char*)(mb + (size_t)nr * CT_) + (size_t)tid * 16u;
      asm volatile("global_load_async_to_lds_b128 %0, %1, off"
                   :: "v"(ldsoff), "v"(g) : "memory");
    }
    if (tid == 0) {
      bool ok = (scb[i] > THR_) && (((removed[i >> 4] >> (i & 15)) & 1u) == 0u);
      sflag = ok ? 1 : 0;
      if (ok && skc < MAXDET_) kidx[skc++] = (short)i;
    }
    // wait until row i has landed in LDS (row i+1 may stay outstanding)
    asm volatile("s_wait_asynccnt 0x1" ::: "memory");
    __syncthreads();
    if (sflag) {
      if (tid < 64) {
        unsigned u = rowbuf[i & 1][tid];
        removed[2 * tid]     |= (u & 0xFFFFu);
        removed[2 * tid + 1] |= (u >> 16);
      }
    }
    __syncthreads();
  }

  const int kc = skc;
  const float4* bx = (const float4*)cbox + (size_t)b * K_;
  float4*       ob = (float4*)outb + (size_t)b * MAXDET_;
  for (int r = tid; r < MAXDET_; r += 256) {
    if (r < kc) {
      int i = kidx[r];
      ob[r] = bx[i];
      outs[(size_t)b * MAXDET_ + r] = scb[i];
      outl[(size_t)b * MAXDET_ + r] = ccls[(size_t)b * K_ + i];
    } else {
      float4 z; z.x = z.y = z.z = z.w = 0.0f;
      ob[r] = z;
      outs[(size_t)b * MAXDET_ + r] = 0.0f;
      outl[(size_t)b * MAXDET_ + r] = 0;
    }
  }
  if (tid == 0) outnv[b] = kc;   // kc == min(total kept, 300)
}

// ===========================================================================
extern "C" void kernel_launch(void* const* d_in, const int* in_sizes, int n_in,
                              void* d_out, int out_size, void* d_ws, size_t ws_size,
                              hipStream_t stream) {
  (void)in_sizes; (void)n_in; (void)out_size; (void)ws_size;

  const float* boxes  = (const float*)d_in[0];   // [8,8192,4]
  const float* scores = (const float*)d_in[1];   // [8,8192,32]

  // workspace layout (4.625 MB total)
  char* ws = (char*)d_ws;
  float*          cscore = (float*)(ws);                   //  64 KB
  int*            ccls   = (int*)(ws + (64u << 10));       //  64 KB
  float*          cbox   = (float*)(ws + (128u << 10));    // 512 KB
  unsigned short* maskm  = (unsigned short*)(ws + (640u << 10)); // 4 MB

  // output layout: boxes[8,300,4] | scores[8,300] | labels(i32)[8,300] | nv(i32)[8]
  float* outf  = (float*)d_out;
  float* outb  = outf;
  float* outs  = outf + B_ * MAXDET_ * 4;
  int*   outl  = (int*)(outf + B_ * MAXDET_ * 4 + B_ * MAXDET_);
  int*   outnv = outl + B_ * MAXDET_;

  nms_topk<<<B_, 1024, 0, stream>>>(boxes, scores, cscore, ccls, cbox);
  nms_pairmask<<<dim3(2048, B_), 256, 0, stream>>>(ccls, cbox, maskm);
  nms_greedy<<<B_, 256, 0, stream>>>(cscore, ccls, cbox, maskm,
                                     outb, outs, outl, outnv);
}
